// DLRM_DHE_20323785245285
// MI455X (gfx1250) — compile-verified, matrix operand order checked by reading, not run on hardware
//
#include <hip/hip_runtime.h>
#include <hip/hip_bf16.h>

typedef __attribute__((ext_vector_type(16))) __bf16 v16bf;
typedef __attribute__((ext_vector_type(8)))  __bf16 v8bf;
typedef __attribute__((ext_vector_type(2)))  __bf16 v2bf;
typedef __attribute__((ext_vector_type(8)))  float  v8f;
typedef __attribute__((ext_vector_type(2)))  float  v2f;

#define NUM_TABLES 26
#define ROWS_T     100000
#define EMB        64
#define BATCH      4096
#define LOOK       32
#define ZCOLS      (EMB * (1 + NUM_TABLES))   // 1728

// ---------------------------------------------------------------------------
// fp32 -> bf16 conversion (weights + dense input)
// ---------------------------------------------------------------------------
__global__ __launch_bounds__(256) void cvt_f32_bf16(const float* __restrict__ in,
                                                    __bf16* __restrict__ out, int n) {
    int i = blockIdx.x * 256 + threadIdx.x;
    if (i < n) out[i] = (__bf16)in[i];
}

// ---------------------------------------------------------------------------
// EmbeddingBag(sum): one wave32 per (table, batch) bag.
// ---------------------------------------------------------------------------
__global__ __launch_bounds__(256) void emb_pool_bags(const float* __restrict__ tables,
                                                     const int*   __restrict__ indices,
                                                     __bf16*      __restrict__ Z) {
    const int gwave = (blockIdx.x * 256 + threadIdx.x) >> 5;   // 0 .. 26*4096-1
    const int lane  = threadIdx.x & 31;
    const int t = gwave >> 12;        // / 4096
    const int b = gwave & 4095;

    const int* idx = indices + ((size_t)t * BATCH + b) * LOOK;
    const int myidx = idx[lane];      // lane l holds index l of the bag
    const float* tbl = tables + (size_t)t * ROWS_T * EMB;

    float sx = 0.f, sy = 0.f;
#pragma unroll 1
    for (int j = 0; j < LOOK; ++j) {
        const int row = __shfl(myidx, j, 32);
        if (j + 2 < LOOK) {
            const int rown = __shfl(myidx, j + 2, 32);
            __builtin_prefetch(tbl + (size_t)rown * EMB + lane * 2, 0, 0);
        }
        v2f v = *(const v2f*)(tbl + (size_t)row * EMB + lane * 2);
        sx += v.x;
        sy += v.y;
    }
    v2bf o;
    o[0] = (__bf16)sx;
    o[1] = (__bf16)sy;
    *(v2bf*)(Z + (size_t)b * ZCOLS + EMB * (1 + t) + lane * 2) = o;
}

// ---------------------------------------------------------------------------
// CDNA5 async global->LDS copy (ASYNCcnt-tracked, no VGPR round trip).
// lds_byte_off is the offset inside the group segment (single static __shared__
// array => segment offset 0). lds_base_escape is an unreferenced operand whose
// only purpose is to make the shared array's address escape into the asm so
// alias analysis keeps the subsequent ds_load reads alive.
// ---------------------------------------------------------------------------
__device__ __forceinline__ void async_to_lds_b128(unsigned lds_byte_off,
                                                  unsigned long long gaddr,
                                                  void* lds_base_escape) {
    asm volatile("global_load_async_to_lds_b128 %0, %1, off"
                 :: "v"(lds_byte_off), "v"(gaddr), "v"(lds_base_escape)
                 : "memory");
}
__device__ __forceinline__ void wait_async_le1() {
    asm volatile("s_wait_asynccnt 1" ::: "memory");
}
__device__ __forceinline__ void wait_async_le0() {
    asm volatile("s_wait_asynccnt 0" ::: "memory");
}

// ---------------------------------------------------------------------------
// WMMA 16x16x32 bf16 fragment from global (A operand).
// ISA 16-bit layout: lane L -> row (L&15), half (L>>4);
//   elems 0..7  = K[half*8 .. half*8+7], elems 8..15 = K[16+half*8 ..]
// ---------------------------------------------------------------------------
__device__ __forceinline__ v16bf load_frag16x32(const __bf16* __restrict__ base,
                                                int ld, int k0) {
    const int lane = threadIdx.x & 31;
    const int r    = lane & 15;
    const int half = lane >> 4;
    const __bf16* p = base + (size_t)r * ld + k0 + half * 8;
    v8bf lo = *(const v8bf*)(p);
    v8bf hi = *(const v8bf*)(p + 16);
    v16bf f;
#pragma unroll
    for (int i = 0; i < 8; ++i) { f[i] = lo[i]; f[i + 8] = hi[i]; }
    return f;
}

// B fragment out of the LDS-staged 64x32 bf16 weight tile (row-major, 64B rows).
__device__ __forceinline__ v16bf frag_from_lds(const __bf16* sbase, int nt) {
    const int lane = threadIdx.x & 31;
    const int r    = lane & 15;
    const int half = lane >> 4;
    const __bf16* p = sbase + (nt * 16 + r) * 32 + half * 8;
    v8bf lo = *(const v8bf*)(p);
    v8bf hi = *(const v8bf*)(p + 16);
    v16bf f;
#pragma unroll
    for (int i = 0; i < 8; ++i) { f[i] = lo[i]; f[i + 8] = hi[i]; }
    return f;
}

// ---------------------------------------------------------------------------
// C[M,N] = relu(A[M,K] @ W[N,K]^T + bias), bf16 in / f32 accum / bf16 out.
// Block = 256 threads = 8 waves; wave w owns rows [by*128 + w*16, +16),
// block owns cols [bx*64, +64). Weight tile (64 rows x 32 K, 4KB) is staged
// through LDS with double-buffered async copies: 8 async b128 per block per
// K-step instead of 64 redundant global_load_b128. Last iteration peeled so
// the steady-state body is branch-free.
// ---------------------------------------------------------------------------
__global__ __launch_bounds__(256) void gemm_bf16_relu(const __bf16* __restrict__ A, int lda,
                                                      const __bf16* __restrict__ W, int ldw,
                                                      const float*  __restrict__ bias,
                                                      __bf16* __restrict__ C, int ldc,
                                                      int K) {
    __shared__ __align__(16) __bf16 sW[2][64][32];   // 2 x 4KB double buffer

    const int tid  = threadIdx.x;
    const int lane = tid & 31;
    const int wave = tid >> 5;
    const int mrow = blockIdx.y * 128 + wave * 16;
    const int ncol = blockIdx.x * 64;

    // This thread's slice of the async weight-tile copy: 256 threads x 16B = 4KB
    const int srow = tid >> 2;                 // 0..63 (weight row within tile)
    const int schk = tid & 3;                  // 0..3  (16B chunk within row)
    const unsigned long long gW =
        (unsigned long long)(const void*)(W + (size_t)(ncol + srow) * ldw + schk * 8);
    const unsigned ldsoff = (unsigned)(srow * 64 + schk * 16);   // bytes in buffer
    void* sW_escape = (void*)&sW[0][0][0];

    const __bf16* Ab = A + (size_t)mrow * lda;

    v8f acc[4] = {v8f{}, v8f{}, v8f{}, v8f{}};

    const int nsteps = K >> 5;                 // >= 1, K multiple of 32

    // Prologue: stage K-tile 0 into buffer 0
    async_to_lds_b128(ldsoff, gW, sW_escape);

    for (int s = 0; s < nsteps - 1; ++s) {
        const int cur = s & 1;
        // A fragment for this step: issue the global load early so it overlaps
        // the async wait + barrier.
        v16bf a = load_frag16x32(Ab, lda, s << 5);
        // Stage next K-tile into the other buffer while we compute this one.
        async_to_lds_b128((unsigned)((1 - cur) * 4096) + ldsoff,
                          gW + (unsigned long long)(s + 1) * 64u, sW_escape);
        wait_async_le1();              // this wave's tile s has landed
        __syncthreads();               // all waves' slices of tile s visible

        const __bf16* sb = &sW[cur][0][0];
        v16bf b0 = frag_from_lds(sb, 0);
        v16bf b1 = frag_from_lds(sb, 1);
        v16bf b2 = frag_from_lds(sb, 2);
        v16bf b3 = frag_from_lds(sb, 3);
        acc[0] = __builtin_amdgcn_wmma_f32_16x16x32_bf16(false, a, false, b0, (short)0, acc[0], false, false);
        acc[1] = __builtin_amdgcn_wmma_f32_16x16x32_bf16(false, a, false, b1, (short)0, acc[1], false, false);
        acc[2] = __builtin_amdgcn_wmma_f32_16x16x32_bf16(false, a, false, b2, (short)0, acc[2], false, false);
        acc[3] = __builtin_amdgcn_wmma_f32_16x16x32_bf16(false, a, false, b3, (short)0, acc[3], false, false);

        __syncthreads();               // tile s fully consumed before its buffer
                                       // is re-targeted by the next issue
    }

    // Peeled last step
    {
        const int s = nsteps - 1;
        const int cur = s & 1;
        v16bf a = load_frag16x32(Ab, lda, s << 5);
        wait_async_le0();
        __syncthreads();
        const __bf16* sb = &sW[cur][0][0];
        v16bf b0 = frag_from_lds(sb, 0);
        v16bf b1 = frag_from_lds(sb, 1);
        v16bf b2 = frag_from_lds(sb, 2);
        v16bf b3 = frag_from_lds(sb, 3);
        acc[0] = __builtin_amdgcn_wmma_f32_16x16x32_bf16(false, a, false, b0, (short)0, acc[0], false, false);
        acc[1] = __builtin_amdgcn_wmma_f32_16x16x32_bf16(false, a, false, b1, (short)0, acc[1], false, false);
        acc[2] = __builtin_amdgcn_wmma_f32_16x16x32_bf16(false, a, false, b2, (short)0, acc[2], false, false);
        acc[3] = __builtin_amdgcn_wmma_f32_16x16x32_bf16(false, a, false, b3, (short)0, acc[3], false, false);
    }

    // C/D layout: lanes 0-15 -> N=lane, M=vgpr r; lanes 16-31 -> N=lane-16, M=8+r
    const int nloc  = lane & 15;
    const int rbase = (lane >> 4) * 8;
#pragma unroll
    for (int nt = 0; nt < 4; ++nt) {
        const int col = ncol + nt * 16 + nloc;
        const float bv = bias[col];
#pragma unroll
        for (int r = 0; r < 8; ++r) {
            float v = acc[nt][r] + bv;
            v = v > 0.f ? v : 0.f;
            C[(size_t)(mrow + rbase + r) * ldc + col] = (__bf16)v;
        }
    }
}

// ---------------------------------------------------------------------------
// Final layer: out[b] = sigmoid(dot(U2[b, 0:256], w2[0:256]) + b2).
// ---------------------------------------------------------------------------
__global__ __launch_bounds__(256) void top_final(const __bf16* __restrict__ U2,
                                                 const float*  __restrict__ w2,
                                                 const float*  __restrict__ b2,
                                                 float* __restrict__ out) {
    const int row  = (blockIdx.x * 256 + threadIdx.x) >> 5;
    const int lane = threadIdx.x & 31;
    const __bf16* u = U2 + (size_t)row * 256 + lane * 8;
    const float*  w = w2 + lane * 8;
    float s = 0.f;
#pragma unroll
    for (int i = 0; i < 8; ++i) s += (float)u[i] * w[i];
#pragma unroll
    for (int off = 16; off > 0; off >>= 1) s += __shfl_xor(s, off, 32);
    if (lane == 0) {
        const float z = s + b2[0];
        out[row] = 1.f / (1.f + __expf(-z));
    }
}

// ---------------------------------------------------------------------------
extern "C" void kernel_launch(void* const* d_in, const int* in_sizes, int n_in,
                              void* d_out, int out_size, void* d_ws, size_t ws_size,
                              hipStream_t stream) {
    const float* x_dense = (const float*)d_in[0];
    const int*   x_idx   = (const int*)  d_in[1];
    const float* emb_t   = (const float*)d_in[2];
    const float* bw0 = (const float*)d_in[3];  const float* bb0 = (const float*)d_in[4];
    const float* bw1 = (const float*)d_in[5];  const float* bb1 = (const float*)d_in[6];
    const float* bw2 = (const float*)d_in[7];  const float* bb2 = (const float*)d_in[8];
    const float* tw0 = (const float*)d_in[9];  const float* tb0 = (const float*)d_in[10];
    const float* tw1 = (const float*)d_in[11]; const float* tb1 = (const float*)d_in[12];
    const float* tw2 = (const float*)d_in[13]; const float* tb2 = (const float*)d_in[14];
    float* out = (float*)d_out;

    char* ws = (char*)d_ws;
    size_t off = 0;
    auto alloc_bf16 = [&](size_t elems) -> __bf16* {
        off = (off + 255) & ~(size_t)255;
        __bf16* p = (__bf16*)(ws + off);
        off += elems * sizeof(__bf16);
        return p;
    };

    __bf16* Xbf = alloc_bf16((size_t)BATCH * 256);        // dense input, bf16
    __bf16* Wb0 = alloc_bf16((size_t)512 * 256);
    __bf16* Wb1 = alloc_bf16((size_t)256 * 512);
    __bf16* Wb2 = alloc_bf16((size_t)64  * 256);
    __bf16* Wt0 = alloc_bf16((size_t)512 * ZCOLS);
    __bf16* Wt1 = alloc_bf16((size_t)256 * 512);
    __bf16* H1  = alloc_bf16((size_t)BATCH * 512);
    __bf16* H2  = alloc_bf16((size_t)BATCH * 256);
    __bf16* Z   = alloc_bf16((size_t)BATCH * ZCOLS);      // [h | emb] concat
    __bf16* U1  = alloc_bf16((size_t)BATCH * 512);
    __bf16* U2  = alloc_bf16((size_t)BATCH * 256);
    (void)ws_size; (void)in_sizes; (void)n_in; (void)out_size;

    const dim3 blk(256);
    auto cvt = [&](const float* src, __bf16* dst, int n) {
        cvt_f32_bf16<<<dim3((n + 255) / 256), blk, 0, stream>>>(src, dst, n);
    };
    cvt(x_dense, Xbf, BATCH * 256);
    cvt(bw0, Wb0, 512 * 256);
    cvt(bw1, Wb1, 256 * 512);
    cvt(bw2, Wb2, 64 * 256);
    cvt(tw0, Wt0, 512 * ZCOLS);
    cvt(tw1, Wt1, 256 * 512);

    // Embedding pooling -> Z[:, 64:1728]
    emb_pool_bags<<<dim3(NUM_TABLES * BATCH / 8), blk, 0, stream>>>(emb_t, x_idx, Z);

    // Bottom MLP: 256 -> 512 -> 256 -> 64 (ReLU each), last writes Z[:, 0:64]
    gemm_bf16_relu<<<dim3(512 / 64, BATCH / 128), blk, 0, stream>>>(Xbf, 256, Wb0, 256, bb0, H1, 512, 256);
    gemm_bf16_relu<<<dim3(256 / 64, BATCH / 128), blk, 0, stream>>>(H1, 512, Wb1, 512, bb1, H2, 256, 512);
    gemm_bf16_relu<<<dim3(64  / 64, BATCH / 128), blk, 0, stream>>>(H2, 256, Wb2, 256, bb2, Z, ZCOLS, 256);

    // Top MLP: 1728 -> 512 -> 256 (ReLU), then 256 -> 1 + sigmoid
    gemm_bf16_relu<<<dim3(512 / 64, BATCH / 128), blk, 0, stream>>>(Z, ZCOLS, Wt0, ZCOLS, tb0, U1, 512, ZCOLS);
    gemm_bf16_relu<<<dim3(256 / 64, BATCH / 128), blk, 0, stream>>>(U1, 512, Wt1, 512, tb1, U2, 256, 512);
    top_final<<<dim3(BATCH / 8), blk, 0, stream>>>(U2, tw2, tb2, out);
}
